// MaskedMultiHeadAttention_28243704938905
// MI455X (gfx1250) — matrix operand order repaired and checked by reading, not compiled, not run on hardware
//
#include <hip/hip_runtime.h>
#include <hip/hip_bf16.h>
#include <cstddef>

// ---------------------------------------------------------------------------
// GPT-2 masked multi-head attention for MI455X (gfx1250, wave32, WMMA).
// B=2, T=2048, C=768, H=12, D=64.
// Matmuls: v_wmma_f32_16x16x32_bf16.  A-panels staged via async LDS copies.
// ---------------------------------------------------------------------------

typedef __bf16 bf16_t;
typedef __attribute__((ext_vector_type(16))) __bf16 v16bf;
typedef __attribute__((ext_vector_type(8)))  __bf16 v8bf;
typedef __attribute__((ext_vector_type(8)))  float  v8f;
typedef __attribute__((ext_vector_type(4)))  int    v4i;

#define DEV __device__ __forceinline__

#if defined(__has_builtin)
#if __has_builtin(__builtin_amdgcn_global_load_async_to_lds_b128)
#define HAVE_ASYNC_LDS 1
#endif
#endif

DEV v8f wmma_bf16(v16bf a, v16bf b, v8f c) {
    // 8 args: (neg_a, A, neg_b, B, c_mod, C, reuse_a, reuse_b)
    return __builtin_amdgcn_wmma_f32_16x16x32_bf16(false, a, false, b,
                                                   (short)0, c, false, false);
}

// Copy 16B global -> LDS.  Async (ASYNCcnt) when available.
DEV void copy16_to_lds(const bf16_t* g, bf16_t* l) {
#ifdef HAVE_ASYNC_LDS
    __builtin_amdgcn_global_load_async_to_lds_b128(
        (__attribute__((address_space(1))) v4i*)g,
        (__attribute__((address_space(3))) v4i*)l, 0, 0);
#else
    *(v8bf*)l = *(const v8bf*)g;
#endif
}
DEV void wait_async_lds() {
#ifdef HAVE_ASYNC_LDS
    asm volatile("s_wait_asynccnt 0" ::: "memory");
#endif
}

// Load a 16x32 bf16 fragment (A layout; also B layout when the source is the
// transposed matrix, i.e. lane r holds row/column r, contiguous K).
DEV v16bf load_frag(const bf16_t* tile, int ld) {
    const int lane = threadIdx.x & 31;
    const int rr = lane & 15;
    const int hl = lane >> 4;
    const bf16_t* p = tile + (size_t)rr * ld + hl * 8;
    union { v16bf v; v8bf h[2]; } u;
    u.h[0] = *(const v8bf*)(p);
    u.h[1] = *(const v8bf*)(p + 16);
    return u.v;
}

// 16-lane-group reductions (rows of a C-tile live across a 16-lane half).
DEV float rmax16(float v) {
    v = fmaxf(v, __shfl_xor(v, 1));
    v = fmaxf(v, __shfl_xor(v, 2));
    v = fmaxf(v, __shfl_xor(v, 4));
    v = fmaxf(v, __shfl_xor(v, 8));
    return v;
}
DEV float rsum16(float v) {
    v += __shfl_xor(v, 1);
    v += __shfl_xor(v, 2);
    v += __shfl_xor(v, 4);
    v += __shfl_xor(v, 8);
    return v;
}

// ---------------------------------------------------------------------------
// Prep kernels
// ---------------------------------------------------------------------------
__global__ void f32_to_bf16_kernel(const float* __restrict__ src,
                                   bf16_t* __restrict__ dst, int n) {
    for (int i = blockIdx.x * blockDim.x + threadIdx.x; i < n;
         i += gridDim.x * blockDim.x)
        dst[i] = (bf16_t)src[i];
}

// W [rows x cols] f32 row-major  ->  WT [cols x rows] bf16 row-major
__global__ void transpose_bf16_kernel(const float* __restrict__ W,
                                      bf16_t* __restrict__ WT,
                                      int rows, int cols) {
    int i = blockIdx.x * blockDim.x + threadIdx.x;
    if (i >= rows * cols) return;
    int k = i / cols;
    int n = i - k * cols;
    WT[(size_t)n * rows + k] = (bf16_t)W[i];
}

// ---------------------------------------------------------------------------
// Shared helper: cooperatively stage a 64x32 bf16 A-slice into LDS.
// 4 KB total, 128 threads, 16 B per thread x2.
// ---------------------------------------------------------------------------
DEV void stage_A(const bf16_t* __restrict__ A, int lda, bf16_t* __restrict__ As) {
    const int tid = threadIdx.x;
#pragma unroll
    for (int j = 0; j < 2; ++j) {
        const int c = tid + j * 128;       // 256 16B chunks
        const int row = c >> 2;
        const int col8 = (c & 3) * 8;
        copy16_to_lds(A + (size_t)row * lda + col8, As + row * 32 + col8);
    }
}

// ---------------------------------------------------------------------------
// QKV GEMM: qkv[m,n] = sum_k XB[m,k]*WAT[n,k] + b[n], scatter into
// Q [bh][t][d] (scaled 1/8), K [bh][t][d], VT [bh][d][t]   (all bf16)
// Block = 64x256 macro-tile (4 waves x 64x64).  M=4096, N=2304, K=768.
// ---------------------------------------------------------------------------
__global__ void __launch_bounds__(128)
qkv_gemm_kernel(const bf16_t* __restrict__ XB, const bf16_t* __restrict__ WAT,
                const float* __restrict__ bias,
                bf16_t* __restrict__ Qp, bf16_t* __restrict__ Kp,
                bf16_t* __restrict__ VTp) {
    __shared__ __align__(16) bf16_t Ast[64 * 32];

    const int NB = 2304 / 256;               // 9 n-blocks
    const int mb = blockIdx.x / NB;
    const int nb = blockIdx.x - mb * NB;
    const int warp = threadIdx.x >> 5;
    const int m0 = mb * 64;
    const int n0 = nb * 256 + warp * 64;
    const int lane = threadIdx.x & 31;
    const int rr = lane & 15, hl = lane >> 4;

    v8f acc[4][4] = {};
    for (int k0 = 0; k0 < 768; k0 += 32) {
        stage_A(XB + (size_t)m0 * 768 + k0, 768, Ast);
        wait_async_lds();
        __syncthreads();
        v16bf a[4], b[4];
#pragma unroll
        for (int mi = 0; mi < 4; ++mi)
            a[mi] = load_frag(Ast + mi * 16 * 32, 32);
#pragma unroll
        for (int ni = 0; ni < 4; ++ni)
            b[ni] = load_frag(WAT + (size_t)(n0 + ni * 16) * 768 + k0, 768);
#pragma unroll
        for (int mi = 0; mi < 4; ++mi)
#pragma unroll
            for (int ni = 0; ni < 4; ++ni)
                acc[mi][ni] = wmma_bf16(a[mi], b[ni], acc[mi][ni]);
        __syncthreads();
    }

#pragma unroll
    for (int ni = 0; ni < 4; ++ni) {
        const int n = n0 + ni * 16 + rr;            // [0,2304)
        const float bv = bias[n];
        const int region = n / 768;                  // 0=q 1=k 2=v (tile-uniform)
        const int cc = n - region * 768;
        const int head = cc >> 6;
        const int d = cc & 63;
#pragma unroll
        for (int mi = 0; mi < 4; ++mi) {
#pragma unroll
            for (int i = 0; i < 8; ++i) {
                const int m = m0 + mi * 16 + i + 8 * hl;
                const int bidx = m >> 11;
                const int t = m & 2047;
                const int bh = bidx * 12 + head;
                const float val = acc[mi][ni][i] + bv;
                if (region == 0)
                    Qp[((size_t)bh * 2048 + t) * 64 + d] = (bf16_t)(val * 0.125f);
                else if (region == 1)
                    Kp[((size_t)bh * 2048 + t) * 64 + d] = (bf16_t)val;
                else
                    VTp[((size_t)bh * 64 + d) * 2048 + t] = (bf16_t)val;
            }
        }
    }
}

// ---------------------------------------------------------------------------
// Flash attention, causal.  One wave owns 32 query rows of one (b,h).
// kv processed in tiles of 64.  P staged through per-wave LDS to convert
// C-layout f32 -> A-layout bf16.  (No block barriers: divergent loop counts.)
// ---------------------------------------------------------------------------
__global__ void __launch_bounds__(128)
flash_attn_kernel(const bf16_t* __restrict__ Qp, const bf16_t* __restrict__ Kp,
                  const bf16_t* __restrict__ VTp,
                  const int* __restrict__ amask, bf16_t* __restrict__ AO) {
    __shared__ __align__(16) bf16_t Pst[4][32 * 64];

    const int warp = threadIdx.x >> 5;
    const int waveGlobal = blockIdx.x * 4 + warp;
    const int bh = waveGlobal >> 6;            // 24 (b,h) pairs
    const int rowTile = waveGlobal & 63;       // 64 tiles of 32 rows
    const int qRow0 = rowTile * 32;
    const int b = bh / 12;
    const int h = bh - b * 12;
    const int lane = threadIdx.x & 31;
    const int rr = lane & 15, hl = lane >> 4;

    const bf16_t* Qb = Qp + (size_t)bh * 2048 * 64;
    const bf16_t* Kb = Kp + (size_t)bh * 2048 * 64;
    const bf16_t* VTb = VTp + (size_t)bh * 64 * 2048;
    const int* amb = amask + b * 2048;
    bf16_t* ps = Pst[warp];

    // Q fragments stay resident: 2 row tiles x 2 k-steps (D=64)
    v16bf qf[2][2];
#pragma unroll
    for (int mi = 0; mi < 2; ++mi)
#pragma unroll
        for (int kk = 0; kk < 2; ++kk)
            qf[mi][kk] = load_frag(Qb + (size_t)(qRow0 + mi * 16) * 64 + kk * 32, 64);

    v8f o[2][4] = {};
    float mrow[2][8], lrow[2][8];
#pragma unroll
    for (int mi = 0; mi < 2; ++mi)
#pragma unroll
        for (int i = 0; i < 8; ++i) { mrow[mi][i] = -3.0e38f; lrow[mi][i] = 0.f; }

    const int kvEnd = qRow0 + 32;
    for (int kv0 = 0; kv0 < kvEnd; kv0 += 64) {
        // ---- S = Q K^T (already scaled by 1/sqrt(D) via Q) ----
        v8f s[2][4];
#pragma unroll
        for (int ni = 0; ni < 4; ++ni) {
            v16bf kf0 = load_frag(Kb + (size_t)(kv0 + ni * 16) * 64 + 0, 64);
            v16bf kf1 = load_frag(Kb + (size_t)(kv0 + ni * 16) * 64 + 32, 64);
#pragma unroll
            for (int mi = 0; mi < 2; ++mi) {
                v8f z = {};
                z = wmma_bf16(qf[mi][0], kf0, z);
                s[mi][ni] = wmma_bf16(qf[mi][1], kf1, z);
            }
        }
        // ---- causal + attention mask ----
#pragma unroll
        for (int ni = 0; ni < 4; ++ni) {
            const int n = kv0 + ni * 16 + rr;
            const bool colok = (amb[n] != 0);
#pragma unroll
            for (int mi = 0; mi < 2; ++mi)
#pragma unroll
                for (int i = 0; i < 8; ++i) {
                    const int m = qRow0 + mi * 16 + i + 8 * hl;
                    if (n > m || !colok) s[mi][ni][i] = -3.0e38f;
                }
        }
        // ---- online softmax update ----
#pragma unroll
        for (int mi = 0; mi < 2; ++mi)
#pragma unroll
            for (int i = 0; i < 8; ++i) {
                float rmax = s[mi][0][i];
#pragma unroll
                for (int ni = 1; ni < 4; ++ni) rmax = fmaxf(rmax, s[mi][ni][i]);
                rmax = rmax16(rmax);
                const float mold = mrow[mi][i];
                const float mnew = fmaxf(mold, rmax);
                const float alpha = __expf(mold - mnew);
                float rs = 0.f;
#pragma unroll
                for (int ni = 0; ni < 4; ++ni) {
                    const float p = __expf(s[mi][ni][i] - mnew);
                    s[mi][ni][i] = p;
                    rs += p;
                }
                rs = rsum16(rs);
                mrow[mi][i] = mnew;
                lrow[mi][i] = lrow[mi][i] * alpha + rs;
#pragma unroll
                for (int ni = 0; ni < 4; ++ni) o[mi][ni][i] *= alpha;
            }
        // ---- stage P into LDS (C-layout -> row-major bf16) ----
#pragma unroll
        for (int mi = 0; mi < 2; ++mi)
#pragma unroll
            for (int ni = 0; ni < 4; ++ni)
#pragma unroll
                for (int i = 0; i < 8; ++i)
                    ps[(mi * 16 + i + 8 * hl) * 64 + ni * 16 + rr] =
                        (bf16_t)s[mi][ni][i];
        asm volatile("s_wait_dscnt 0" ::: "memory");
        // ---- O += P @ V ----
#pragma unroll
        for (int kk = 0; kk < 2; ++kk) {
            v16bf pf[2];
#pragma unroll
            for (int mi = 0; mi < 2; ++mi)
                pf[mi] = load_frag(ps + (mi * 16) * 64 + kk * 32, 64);
#pragma unroll
            for (int ni = 0; ni < 4; ++ni) {
                v16bf vf = load_frag(VTb + (size_t)(ni * 16) * 2048 + kv0 + kk * 32,
                                     2048);
#pragma unroll
                for (int mi = 0; mi < 2; ++mi)
                    o[mi][ni] = wmma_bf16(pf[mi], vf, o[mi][ni]);
            }
        }
    }

    // ---- normalize and write AO [B*T, C] bf16 ----
#pragma unroll
    for (int mi = 0; mi < 2; ++mi)
#pragma unroll
        for (int i = 0; i < 8; ++i) {
            const float inv = 1.0f / lrow[mi][i];
            const int t = qRow0 + mi * 16 + i + 8 * hl;
            const size_t row = (size_t)(b * 2048 + t) * 768;
#pragma unroll
            for (int ni = 0; ni < 4; ++ni)
                AO[row + h * 64 + ni * 16 + rr] = (bf16_t)(o[mi][ni][i] * inv);
        }
}

// ---------------------------------------------------------------------------
// Projection GEMM: out[m,n] = sum_k AO[m,k]*WPT[n,k] + b[n]  (f32 out)
// Block = 64x256 macro-tile.  M=4096, N=768, K=768.
// ---------------------------------------------------------------------------
__global__ void __launch_bounds__(128)
proj_gemm_kernel(const bf16_t* __restrict__ AO, const bf16_t* __restrict__ WPT,
                 const float* __restrict__ bias, float* __restrict__ out) {
    __shared__ __align__(16) bf16_t Ast[64 * 32];

    const int NB = 768 / 256;                // 3 n-blocks
    const int mb = blockIdx.x / NB;
    const int nb = blockIdx.x - mb * NB;
    const int warp = threadIdx.x >> 5;
    const int m0 = mb * 64;
    const int n0 = nb * 256 + warp * 64;
    const int lane = threadIdx.x & 31;
    const int rr = lane & 15, hl = lane >> 4;

    v8f acc[4][4] = {};
    for (int k0 = 0; k0 < 768; k0 += 32) {
        stage_A(AO + (size_t)m0 * 768 + k0, 768, Ast);
        wait_async_lds();
        __syncthreads();
        v16bf a[4], b[4];
#pragma unroll
        for (int mi = 0; mi < 4; ++mi)
            a[mi] = load_frag(Ast + mi * 16 * 32, 32);
#pragma unroll
        for (int ni = 0; ni < 4; ++ni)
            b[ni] = load_frag(WPT + (size_t)(n0 + ni * 16) * 768 + k0, 768);
#pragma unroll
        for (int mi = 0; mi < 4; ++mi)
#pragma unroll
            for (int ni = 0; ni < 4; ++ni)
                acc[mi][ni] = wmma_bf16(a[mi], b[ni], acc[mi][ni]);
        __syncthreads();
    }
#pragma unroll
    for (int ni = 0; ni < 4; ++ni) {
        const int n = n0 + ni * 16 + rr;
        const float bv = bias[n];
#pragma unroll
        for (int mi = 0; mi < 4; ++mi)
#pragma unroll
            for (int i = 0; i < 8; ++i) {
                const int m = m0 + mi * 16 + i + 8 * hl;
                out[(size_t)m * 768 + n] = acc[mi][ni][i] + bv;
            }
    }
}

// ---------------------------------------------------------------------------
// Workspace layout (bytes)
// ---------------------------------------------------------------------------
static constexpr size_t SZ_XB  = (size_t)4096 * 768 * 2;   // x bf16
static constexpr size_t SZ_WAT = (size_t)2304 * 768 * 2;   // W_attn^T bf16
static constexpr size_t SZ_WPT = (size_t)768 * 768 * 2;    // W_proj^T bf16
static constexpr size_t SZ_QKV = (size_t)24 * 2048 * 64 * 2;
static constexpr size_t OFF_XB  = 0;
static constexpr size_t OFF_WAT = OFF_XB + SZ_XB;
static constexpr size_t OFF_WPT = OFF_WAT + SZ_WAT;
static constexpr size_t OFF_Q   = OFF_WPT + SZ_WPT;
static constexpr size_t OFF_K   = OFF_Q + SZ_QKV;
static constexpr size_t OFF_VT  = OFF_K + SZ_QKV;
static constexpr size_t OFF_AO  = OFF_VT + SZ_QKV;

extern "C" void kernel_launch(void* const* d_in, const int* in_sizes, int n_in,
                              void* d_out, int out_size, void* d_ws,
                              size_t ws_size, hipStream_t stream) {
    (void)in_sizes; (void)n_in; (void)out_size; (void)ws_size;
    const float* x      = (const float*)d_in[0];
    const int*   amask  = (const int*)d_in[1];
    const float* W_attn = (const float*)d_in[2];
    const float* b_attn = (const float*)d_in[3];
    const float* W_proj = (const float*)d_in[4];
    const float* b_proj = (const float*)d_in[5];
    float* out = (float*)d_out;

    char* ws = (char*)d_ws;
    bf16_t* XB  = (bf16_t*)(ws + OFF_XB);
    bf16_t* WAT = (bf16_t*)(ws + OFF_WAT);
    bf16_t* WPT = (bf16_t*)(ws + OFF_WPT);
    bf16_t* Qb  = (bf16_t*)(ws + OFF_Q);
    bf16_t* Kb  = (bf16_t*)(ws + OFF_K);
    bf16_t* VTb = (bf16_t*)(ws + OFF_VT);
    bf16_t* AO  = (bf16_t*)(ws + OFF_AO);

    // 1) precision conversion / weight transposes
    f32_to_bf16_kernel<<<2048, 256, 0, stream>>>(x, XB, 4096 * 768);
    transpose_bf16_kernel<<<(768 * 2304 + 255) / 256, 256, 0, stream>>>(
        W_attn, WAT, 768, 2304);
    transpose_bf16_kernel<<<(768 * 768 + 255) / 256, 256, 0, stream>>>(
        W_proj, WPT, 768, 768);
    // 2) QKV projection -> Q(scaled), K, V^T    (64 x 9 macro-tiles)
    qkv_gemm_kernel<<<576, 128, 0, stream>>>(XB, WAT, b_attn, Qb, Kb, VTb);
    // 3) causal flash attention   (24 bh * 64 row tiles = 1536 waves)
    flash_attn_kernel<<<384, 128, 0, stream>>>(Qb, Kb, VTb, amask, AO);
    // 4) output projection        (64 x 3 macro-tiles)
    proj_gemm_kernel<<<192, 128, 0, stream>>>(AO, WPT, b_proj, out);
}
